// ResidualSparseSkillMoEFFN_44341242364493
// MI455X (gfx1250) — matrix operand order repaired and checked by reading, not compiled, not run on hardware
//
#include <hip/hip_runtime.h>
#include <math.h>

typedef _Float16 f16;
typedef __attribute__((ext_vector_type(8)))  _Float16 v8h;
typedef __attribute__((ext_vector_type(16))) _Float16 v16h;
typedef __attribute__((ext_vector_type(8)))  float    v8f;

#define T_TOK 16384
#define HD    1024
#define DFFD  4096
#define EHD   512
#define NE    8
#define ALPHAF 0.5f
#define BK    64   // K-chunk per LDS stage
#define LDT   72   // padded LDS row stride (elems); 144B = 9*16B keeps b128 alignment

// ---------------- small helpers ----------------
__device__ __forceinline__ v8f v8f_zero() {
  v8f z;
#pragma unroll
  for (int i = 0; i < 8; ++i) z[i] = 0.f;
  return z;
}
__device__ __forceinline__ v8h v8h_zero() {
  v8h z;
#pragma unroll
  for (int i = 0; i < 8; ++i) z[i] = (f16)0.f;
  return z;
}

__device__ __forceinline__ v8f wmma16(v16h a, v16h b, v8f c) {
  return __builtin_amdgcn_wmma_f32_16x16x32_f16(false, a, false, b, (short)0, c, false, false);
}

// async copy of one 16B chunk: global -> LDS (tracked by ASYNCcnt)
__device__ __forceinline__ void async_b128(f16* lds_dst, const f16* gsrc) {
  unsigned lds = (unsigned)(unsigned long long)lds_dst;  // addr[31:0] = LDS offset
  asm volatile("global_load_async_to_lds_b128 %0, %1, off"
               :: "v"(lds), "v"(gsrc) : "memory");
}
__device__ __forceinline__ void wait_async_and_sync() {
  asm volatile("s_wait_asynccnt 0x0" ::: "memory");
  __syncthreads();
}

// A fragment (16x32 f16): lane m=lane&15; lo = K[kk+(lane>=16?8:0)..+7], hi = +16
__device__ __forceinline__ v16h frag_a(const f16* sm, int row_base, int kk) {
  int lane = threadIdx.x & 31;
  const f16* p = sm + (size_t)(row_base + (lane & 15)) * LDT + kk + ((lane >> 4) * 8);
  v8h lo = *(const v8h*)p;
  v8h hi = *(const v8h*)(p + 16);
  v16h r;
#pragma unroll
  for (int i = 0; i < 8; ++i) { r[i] = lo[i]; r[i + 8] = hi[i]; }
  return r;
}

// B fragment from W[n][k] rows: lane n=lane&15; lo = K[kk+(lane>=16?8:0)..+7], hi = +16
__device__ __forceinline__ v16h frag_b(const f16* sm, int col_base, int kk) {
  int lane = threadIdx.x & 31;
  const f16* p = sm + (size_t)(col_base + (lane & 15)) * LDT + kk + ((lane >> 4) * 8);
  v8h lo = *(const v8h*)p;
  v8h hi = *(const v8h*)(p + 16);
  v16h r;
#pragma unroll
  for (int i = 0; i < 8; ++i) { r[i] = lo[i]; r[i + 8] = hi[i]; }
  return r;
}

// 128 x BK tile, async global->LDS, all rows valid
__device__ __forceinline__ void tile_async(f16* dst, const f16* __restrict__ src,
                                           int src_ld, size_t row0, int k) {
  for (int i = threadIdx.x; i < 128 * (BK / 8); i += 256) {
    int r = i >> 3, c = (i & 7) << 3;
    async_b128(dst + (size_t)r * LDT + c,
               src + (row0 + (size_t)r) * (size_t)src_ld + k + c);
  }
}

// 128 x BK tile with row validity (zero-fill invalid rows via plain ds store)
__device__ __forceinline__ void tile_async_v(f16* dst, const f16* __restrict__ src,
                                             int src_ld, size_t row0, int k, int valid) {
  for (int i = threadIdx.x; i < 128 * (BK / 8); i += 256) {
    int r = i >> 3, c = (i & 7) << 3;
    f16* d = dst + (size_t)r * LDT + c;
    if (r < valid) async_b128(d, src + (row0 + (size_t)r) * (size_t)src_ld + k + c);
    else           *(v8h*)d = v8h_zero();
  }
}

// indirect rows (token gather)
__device__ __forceinline__ void tile_async_ind(f16* dst, const f16* __restrict__ base,
                                               int src_ld, const int* __restrict__ rowmap,
                                               int nrows, int k) {
  for (int i = threadIdx.x; i < 128 * (BK / 8); i += 256) {
    int r = i >> 3, c = (i & 7) << 3;
    f16* d = dst + (size_t)r * LDT + c;
    if (r < nrows) async_b128(d, base + (size_t)rowmap[r] * (size_t)src_ld + k + c);
    else           *(v8h*)d = v8h_zero();
  }
}

// ---------------- elementwise / routing kernels ----------------
__global__ void k_f32_to_f16(const float* __restrict__ s, f16* __restrict__ d, int n) {
  int i = blockIdx.x * blockDim.x + threadIdx.x;
  if (i < n) d[i] = (f16)s[i];
}

__global__ void k_zero_counts(int* counts) {
  if (threadIdx.x < NE) counts[threadIdx.x] = 0;
}

__global__ void k_router(const float* __restrict__ x, const float* __restrict__ rw,
                         const float* __restrict__ rb, int* __restrict__ eidx,
                         float* __restrict__ gates, int* __restrict__ counts) {
  int gtid = blockIdx.x * blockDim.x + threadIdx.x;
  int t = gtid >> 5;
  int lane = threadIdx.x & 31;
  if (t >= T_TOK) return;
  const float* xr = x + (size_t)t * HD;
  float acc[NE];
#pragma unroll
  for (int e = 0; e < NE; ++e) acc[e] = 0.f;
  for (int c = lane; c < HD; c += 32) {
    float xv = xr[c];
#pragma unroll
    for (int e = 0; e < NE; ++e) acc[e] += xv * rw[e * HD + c];
  }
#pragma unroll
  for (int e = 0; e < NE; ++e) {
    float v = acc[e];
    for (int off = 16; off > 0; off >>= 1) v += __shfl_xor(v, off, 32);
    acc[e] = v;
  }
  if (lane == 0) {
    float l[NE];
#pragma unroll
    for (int e = 0; e < NE; ++e) l[e] = acc[e] + rb[e];
    int i0 = 0;
#pragma unroll
    for (int e = 1; e < NE; ++e) if (l[e] > l[i0]) i0 = e;
    int i1 = -1;
#pragma unroll
    for (int e = 0; e < NE; ++e)
      if (e != i0 && (i1 < 0 || l[e] > l[i1])) i1 = e;
    float g0 = 1.f / (1.f + __expf(l[i1] - l[i0]));  // softmax denom cancels in renorm
    float g1 = 1.f - g0;
    eidx[t * 2] = i0; eidx[t * 2 + 1] = i1;
    gates[t * 2] = g0; gates[t * 2 + 1] = g1;
    atomicAdd(&counts[i0], 1);
    atomicAdd(&counts[i1], 1);
  }
}

__global__ void k_offsets(const int* __restrict__ counts, int* __restrict__ offs,
                          int* __restrict__ cursor) {
  if (threadIdx.x == 0 && blockIdx.x == 0) {
    int s = 0;
    for (int e = 0; e < NE; ++e) { offs[e] = s; s += counts[e]; cursor[e] = 0; }
  }
}

__global__ void k_scatter(const int* __restrict__ eidx, const int* __restrict__ offs,
                          int* __restrict__ cursor, int* __restrict__ bucket_tok,
                          int* __restrict__ pos_of) {
  int t = blockIdx.x * blockDim.x + threadIdx.x;
  if (t >= T_TOK) return;
#pragma unroll
  for (int k = 0; k < 2; ++k) {
    int e = eidx[t * 2 + k];
    int p = offs[e] + atomicAdd(&cursor[e], 1);
    bucket_tok[p] = t;
    pos_of[t * 2 + k] = p;
  }
}

// ---------------- WMMA GEMM kernels ----------------
// Dense fused gate+up+silu: X[T,H] x Wg/Wu[DFF,H]^T -> H[T,DFF] f16
__global__ __launch_bounds__(256) void k_dense_gateup(const f16* __restrict__ X,
                                                      const f16* __restrict__ Wg,
                                                      const f16* __restrict__ Wu,
                                                      f16* __restrict__ Hout) {
  __shared__ f16 sA[128 * LDT], sG[128 * LDT], sU[128 * LDT];
  size_t row0 = (size_t)blockIdx.y * 128;
  size_t col0 = (size_t)blockIdx.x * 128;
  int wave = threadIdx.x >> 5, wm = wave & 3, wn = wave >> 2;
  v8f accg[2][4], accu[2][4];
#pragma unroll
  for (int mt = 0; mt < 2; ++mt)
#pragma unroll
    for (int nt = 0; nt < 4; ++nt) { accg[mt][nt] = v8f_zero(); accu[mt][nt] = v8f_zero(); }

  for (int k = 0; k < HD; k += BK) {
    tile_async(sA, X, HD, row0, k);
    tile_async(sG, Wg, HD, col0, k);
    tile_async(sU, Wu, HD, col0, k);
    wait_async_and_sync();
#pragma unroll
    for (int kk = 0; kk < BK; kk += 32) {
      v16h a[2], bg[4], bu[4];
#pragma unroll
      for (int mt = 0; mt < 2; ++mt) a[mt] = frag_a(sA, wm * 32 + mt * 16, kk);
#pragma unroll
      for (int nt = 0; nt < 4; ++nt) { bg[nt] = frag_b(sG, wn * 64 + nt * 16, kk); bu[nt] = frag_b(sU, wn * 64 + nt * 16, kk); }
#pragma unroll
      for (int mt = 0; mt < 2; ++mt)
#pragma unroll
        for (int nt = 0; nt < 4; ++nt) {
          accg[mt][nt] = wmma16(a[mt], bg[nt], accg[mt][nt]);
          accu[mt][nt] = wmma16(a[mt], bu[nt], accu[mt][nt]);
        }
    }
    __syncthreads();
  }
  int lane = threadIdx.x & 31;
  int lm = (lane >> 4) << 3, ln = lane & 15;
#pragma unroll
  for (int mt = 0; mt < 2; ++mt)
#pragma unroll
    for (int nt = 0; nt < 4; ++nt)
#pragma unroll
      for (int r = 0; r < 8; ++r) {
        float g = accg[mt][nt][r], u = accu[mt][nt][r];
        float h = (g / (1.f + __expf(-g))) * u;  // silu(g)*u
        size_t rr = row0 + wm * 32 + mt * 16 + lm + r;
        size_t cc = col0 + wn * 64 + nt * 16 + ln;
        Hout[rr * DFFD + cc] = (f16)h;
      }
}

// Plain GEMM: A[M,K] x W[N,K]^T -> C[M,N] f32 (full tiles)
__global__ __launch_bounds__(256) void k_gemm_f32(const f16* __restrict__ A,
                                                  const f16* __restrict__ W,
                                                  float* __restrict__ C, int K, int N) {
  __shared__ f16 sA[128 * LDT], sB[128 * LDT];
  size_t row0 = (size_t)blockIdx.y * 128;
  size_t col0 = (size_t)blockIdx.x * 128;
  int wave = threadIdx.x >> 5, wm = wave & 3, wn = wave >> 2;
  v8f acc[2][4];
#pragma unroll
  for (int mt = 0; mt < 2; ++mt)
#pragma unroll
    for (int nt = 0; nt < 4; ++nt) acc[mt][nt] = v8f_zero();
  for (int k = 0; k < K; k += BK) {
    tile_async(sA, A, K, row0, k);
    tile_async(sB, W, K, col0, k);
    wait_async_and_sync();
#pragma unroll
    for (int kk = 0; kk < BK; kk += 32) {
      v16h a[2], b[4];
#pragma unroll
      for (int mt = 0; mt < 2; ++mt) a[mt] = frag_a(sA, wm * 32 + mt * 16, kk);
#pragma unroll
      for (int nt = 0; nt < 4; ++nt) b[nt] = frag_b(sB, wn * 64 + nt * 16, kk);
#pragma unroll
      for (int mt = 0; mt < 2; ++mt)
#pragma unroll
        for (int nt = 0; nt < 4; ++nt) acc[mt][nt] = wmma16(a[mt], b[nt], acc[mt][nt]);
    }
    __syncthreads();
  }
  int lane = threadIdx.x & 31;
  int lm = (lane >> 4) << 3, ln = lane & 15;
#pragma unroll
  for (int mt = 0; mt < 2; ++mt)
#pragma unroll
    for (int nt = 0; nt < 4; ++nt)
#pragma unroll
      for (int r = 0; r < 8; ++r) {
        size_t rr = row0 + wm * 32 + mt * 16 + lm + r;
        size_t cc = col0 + wn * 64 + nt * 16 + ln;
        C[rr * (size_t)N + cc] = acc[mt][nt][r];
      }
}

// Expert fused gate+up+silu with indirect (token-gathered) A rows
__global__ __launch_bounds__(256) void k_exp_gateup(const f16* __restrict__ X,
                                                    const f16* __restrict__ Wg,
                                                    const f16* __restrict__ Wu,
                                                    const int* __restrict__ bucket_tok,
                                                    const int* __restrict__ offs,
                                                    const int* __restrict__ counts,
                                                    f16* __restrict__ Hexp) {
  int e = blockIdx.z;
  int cnt = counts[e];
  int mtile = blockIdx.y;
  if (mtile * 128 >= cnt) return;
  __shared__ f16 sA[128 * LDT], sG[128 * LDT], sU[128 * LDT];
  int seg = offs[e];
  int nrows = cnt - mtile * 128; if (nrows > 128) nrows = 128;
  const int* rowmap = bucket_tok + seg + mtile * 128;
  size_t col0 = (size_t)blockIdx.x * 128;
  const f16* WgE = Wg + (size_t)e * EHD * HD;
  const f16* WuE = Wu + (size_t)e * EHD * HD;
  int wave = threadIdx.x >> 5, wm = wave & 3, wn = wave >> 2;
  v8f accg[2][4], accu[2][4];
#pragma unroll
  for (int mt = 0; mt < 2; ++mt)
#pragma unroll
    for (int nt = 0; nt < 4; ++nt) { accg[mt][nt] = v8f_zero(); accu[mt][nt] = v8f_zero(); }
  for (int k = 0; k < HD; k += BK) {
    tile_async_ind(sA, X, HD, rowmap, nrows, k);
    tile_async(sG, WgE, HD, col0, k);
    tile_async(sU, WuE, HD, col0, k);
    wait_async_and_sync();
#pragma unroll
    for (int kk = 0; kk < BK; kk += 32) {
      v16h a[2], bg[4], bu[4];
#pragma unroll
      for (int mt = 0; mt < 2; ++mt) a[mt] = frag_a(sA, wm * 32 + mt * 16, kk);
#pragma unroll
      for (int nt = 0; nt < 4; ++nt) { bg[nt] = frag_b(sG, wn * 64 + nt * 16, kk); bu[nt] = frag_b(sU, wn * 64 + nt * 16, kk); }
#pragma unroll
      for (int mt = 0; mt < 2; ++mt)
#pragma unroll
        for (int nt = 0; nt < 4; ++nt) {
          accg[mt][nt] = wmma16(a[mt], bg[nt], accg[mt][nt]);
          accu[mt][nt] = wmma16(a[mt], bu[nt], accu[mt][nt]);
        }
    }
    __syncthreads();
  }
  int lane = threadIdx.x & 31;
  int lm = (lane >> 4) << 3, ln = lane & 15;
#pragma unroll
  for (int mt = 0; mt < 2; ++mt)
#pragma unroll
    for (int nt = 0; nt < 4; ++nt)
#pragma unroll
      for (int r = 0; r < 8; ++r) {
        int rl = wm * 32 + mt * 16 + lm + r;
        if (rl < nrows) {
          float g = accg[mt][nt][r], u = accu[mt][nt][r];
          float h = (g / (1.f + __expf(-g))) * u;
          size_t rr = (size_t)(seg + mtile * 128 + rl);
          Hexp[rr * EHD + col0 + wn * 64 + nt * 16 + ln] = (f16)h;
        }
      }
}

// Expert down-proj: Hexp rows x Wd[e][H,EH]^T -> f32 pool rows
__global__ __launch_bounds__(256) void k_exp_down(const f16* __restrict__ Hexp,
                                                  const f16* __restrict__ Wd,
                                                  const int* __restrict__ offs,
                                                  const int* __restrict__ counts,
                                                  float* __restrict__ pool) {
  int e = blockIdx.z;
  int cnt = counts[e];
  int mtile = blockIdx.y;
  if (mtile * 128 >= cnt) return;
  __shared__ f16 sA[128 * LDT], sB[128 * LDT];
  int seg = offs[e];
  int nrows = cnt - mtile * 128; if (nrows > 128) nrows = 128;
  size_t arow0 = (size_t)(seg + mtile * 128);
  size_t col0 = (size_t)blockIdx.x * 128;
  const f16* WdE = Wd + (size_t)e * HD * EHD;
  int wave = threadIdx.x >> 5, wm = wave & 3, wn = wave >> 2;
  v8f acc[2][4];
#pragma unroll
  for (int mt = 0; mt < 2; ++mt)
#pragma unroll
    for (int nt = 0; nt < 4; ++nt) acc[mt][nt] = v8f_zero();
  for (int k = 0; k < EHD; k += BK) {
    tile_async_v(sA, Hexp, EHD, arow0, k, nrows);
    tile_async(sB, WdE, EHD, col0, k);
    wait_async_and_sync();
#pragma unroll
    for (int kk = 0; kk < BK; kk += 32) {
      v16h a[2], b[4];
#pragma unroll
      for (int mt = 0; mt < 2; ++mt) a[mt] = frag_a(sA, wm * 32 + mt * 16, kk);
#pragma unroll
      for (int nt = 0; nt < 4; ++nt) b[nt] = frag_b(sB, wn * 64 + nt * 16, kk);
#pragma unroll
      for (int mt = 0; mt < 2; ++mt)
#pragma unroll
        for (int nt = 0; nt < 4; ++nt) acc[mt][nt] = wmma16(a[mt], b[nt], acc[mt][nt]);
    }
    __syncthreads();
  }
  int lane = threadIdx.x & 31;
  int lm = (lane >> 4) << 3, ln = lane & 15;
#pragma unroll
  for (int mt = 0; mt < 2; ++mt)
#pragma unroll
    for (int nt = 0; nt < 4; ++nt)
#pragma unroll
      for (int r = 0; r < 8; ++r) {
        int rl = wm * 32 + mt * 16 + lm + r;
        if (rl < nrows)
          pool[(arow0 + rl) * HD + col0 + wn * 64 + nt * 16 + ln] = acc[mt][nt][r];
      }
}

// out[t] = dense_out[t] + ALPHA*(g0*pool[p0] + g1*pool[p1])
__global__ void k_combine(float* __restrict__ out, const float* __restrict__ pool,
                          const int* __restrict__ pos_of, const float* __restrict__ gates) {
  int i = blockIdx.x * blockDim.x + threadIdx.x;
  if (i >= T_TOK * (HD / 4)) return;
  int t = i >> 8;           // HD/4 == 256
  int c4 = i & 255;
  int p0 = pos_of[t * 2], p1 = pos_of[t * 2 + 1];
  float g0 = gates[t * 2], g1 = gates[t * 2 + 1];
  const float4* r0 = (const float4*)(pool + (size_t)p0 * HD) + c4;
  const float4* r1 = (const float4*)(pool + (size_t)p1 * HD) + c4;
  float4* o = (float4*)(out + (size_t)t * HD) + c4;
  float4 a = *o, x0 = *r0, x1 = *r1;
  a.x += ALPHAF * (g0 * x0.x + g1 * x1.x);
  a.y += ALPHAF * (g0 * x0.y + g1 * x1.y);
  a.z += ALPHAF * (g0 * x0.z + g1 * x1.z);
  a.w += ALPHAF * (g0 * x0.w + g1 * x1.w);
  *o = a;
}

// ---------------- host launcher ----------------
extern "C" void kernel_launch(void* const* d_in, const int* in_sizes, int n_in,
                              void* d_out, int out_size, void* d_ws, size_t ws_size,
                              hipStream_t stream) {
  const float* x   = (const float*)d_in[0];
  const float* dgw = (const float*)d_in[1];
  const float* duw = (const float*)d_in[2];
  const float* ddw = (const float*)d_in[3];
  const float* rw  = (const float*)d_in[4];
  const float* rb  = (const float*)d_in[5];
  const float* egw = (const float*)d_in[6];
  const float* euw = (const float*)d_in[7];
  const float* edw = (const float*)d_in[8];
  float* out = (float*)d_out;

  char* ws = (char*)d_ws;
  size_t o = 0;
  auto alloc = [&](size_t b) { size_t r = o; o += (b + 255) & ~(size_t)255; return r; };
  f16* xh     = (f16*)(ws + alloc((size_t)T_TOK * HD * 2));
  f16* dgwh   = (f16*)(ws + alloc((size_t)DFFD * HD * 2));
  f16* duwh   = (f16*)(ws + alloc((size_t)DFFD * HD * 2));
  f16* ddwh   = (f16*)(ws + alloc((size_t)HD * DFFD * 2));
  f16* egwh   = (f16*)(ws + alloc((size_t)NE * EHD * HD * 2));
  f16* euwh   = (f16*)(ws + alloc((size_t)NE * EHD * HD * 2));
  f16* edwh   = (f16*)(ws + alloc((size_t)NE * HD * EHD * 2));
  f16* hdense = (f16*)(ws + alloc((size_t)T_TOK * DFFD * 2));
  f16* hexp   = (f16*)(ws + alloc((size_t)2 * T_TOK * EHD * 2));
  float* pool = (float*)(ws + alloc((size_t)2 * T_TOK * HD * 4));
  int* eidx   = (int*)(ws + alloc((size_t)T_TOK * 2 * 4));
  float* gates= (float*)(ws + alloc((size_t)T_TOK * 2 * 4));
  int* pos_of = (int*)(ws + alloc((size_t)T_TOK * 2 * 4));
  int* btok   = (int*)(ws + alloc((size_t)2 * T_TOK * 4));
  int* counts = (int*)(ws + alloc(NE * 4));
  int* offs   = (int*)(ws + alloc(NE * 4));
  int* cursor = (int*)(ws + alloc(NE * 4));

  k_zero_counts<<<1, 32, 0, stream>>>(counts);

  auto cvt = [&](const float* s, f16* d, int n) {
    k_f32_to_f16<<<(n + 255) / 256, 256, 0, stream>>>(s, d, n);
  };
  cvt(x,   xh,   T_TOK * HD);
  cvt(dgw, dgwh, DFFD * HD);
  cvt(duw, duwh, DFFD * HD);
  cvt(ddw, ddwh, HD * DFFD);
  cvt(egw, egwh, NE * EHD * HD);
  cvt(euw, euwh, NE * EHD * HD);
  cvt(edw, edwh, NE * HD * EHD);

  k_router<<<(T_TOK * 32 + 255) / 256, 256, 0, stream>>>(x, rw, rb, eidx, gates, counts);
  k_offsets<<<1, 1, 0, stream>>>(counts, offs, cursor);
  k_scatter<<<(T_TOK + 255) / 256, 256, 0, stream>>>(eidx, offs, cursor, btok, pos_of);

  dim3 g1(DFFD / 128, T_TOK / 128);
  k_dense_gateup<<<g1, 256, 0, stream>>>(xh, dgwh, duwh, hdense);

  dim3 g2(HD / 128, T_TOK / 128);
  k_gemm_f32<<<g2, 256, 0, stream>>>(hdense, ddwh, out, DFFD, HD);

  dim3 g3(EHD / 128, T_TOK / 128, NE);
  k_exp_gateup<<<g3, 256, 0, stream>>>(xh, egwh, euwh, btok, offs, counts, hexp);

  dim3 g4(HD / 128, T_TOK / 128, NE);
  k_exp_down<<<g4, 256, 0, stream>>>(hexp, edwh, offs, counts, pool);

  k_combine<<<(T_TOK * (HD / 4) + 255) / 256, 256, 0, stream>>>(out, pool, pos_of, gates);
}
